// MedLink_83983790506341
// MI455X (gfx1250) — matrix operand order repaired and checked by reading, not compiled, not run on hardware
//
#include <hip/hip_runtime.h>

// MedLink idf-weighted similarity:  scores = (Q .* idf) @ Corpus^T
//   Q=256, C=4096, V=32000, fp32 in/out.
// df pass + fp32->f16 staging, then double-buffered f16 WMMA GEMM (f32 accum).

typedef _Float16 v16h __attribute__((ext_vector_type(16)));
typedef _Float16 v8h  __attribute__((ext_vector_type(8)));
typedef float    v8f  __attribute__((ext_vector_type(8)));

#define NQ 256
#define NC 4096
#define NV 32000

#define BM 64
#define BN 128
#define BK 64
#define LDSTR 72   // 64 data halfs + 8 pad -> row starts 36*m mod 64 banks (conflict-free)

// ---- Kernel 1: zero df counters (ws is poisoned by harness) ----
__global__ void k_zero_df(int* __restrict__ df) {
    int i = blockIdx.x * blockDim.x + threadIdx.x;
    if (i < NV) df[i] = 0;
}

// ---- Kernel 2: df count + corpus fp32 -> f16 staging ----
// grid (NV/256, NC/64), block 256. Coalesced: threads span v, loop over rows.
__global__ void k_df_convert(const float* __restrict__ corpus,
                             _Float16* __restrict__ corpus_h,
                             int* __restrict__ df) {
    int v  = blockIdx.x * 256 + threadIdx.x;   // < NV exactly (125*256)
    int c0 = blockIdx.y * 64;
    int cnt = 0;
    for (int r = 0; r < 64; ++r) {
        size_t idx = (size_t)(c0 + r) * NV + v;
        float x = corpus[idx];
        cnt += (x > 0.0f) ? 1 : 0;
        corpus_h[idx] = (_Float16)x;
    }
    if (cnt) atomicAdd(df + v, cnt);
}

// ---- Kernel 3: queries * idf -> f16 ----
// grid (NV/256, NQ), block 256. idf recomputed per element (2 log1pf, cheap).
__global__ void k_build_q(const float* __restrict__ q,
                          const int* __restrict__ df,
                          _Float16* __restrict__ qh) {
    int v  = blockIdx.x * 256 + threadIdx.x;
    int qi = blockIdx.y;
    float idf = log1pf((float)NC) - log1pf((float)df[v]);
    size_t idx = (size_t)qi * NV + v;
    qh[idx] = (_Float16)(q[idx] * idf);
}

// ---- Kernel 4: double-buffered tiled WMMA GEMM ----
// Block = 128 threads = 4 waves (wave32). Tile: BM=64 x BN=128, K-step 64.
// Wave w computes rows [w*16, w*16+16) x all 128 cols = 8 accumulators.
// Pipeline per iteration: issue global loads for tile i+1 -> for each 32-wide
// K-chunk: preload A + all 8 B fragments into distinct registers, then run the
// 8-WMMA chain back-to-back -> ds_store staged regs to the other buffer ->
// one workgroup barrier. HBM latency hides behind the WMMA chain; LDS latency
// hides behind the fragment-preload batch.
__global__ void __launch_bounds__(128)
k_gemm(const _Float16* __restrict__ qh,
       const _Float16* __restrict__ bh,
       float* __restrict__ out) {
    __shared__ __attribute__((aligned(16))) _Float16 As[2][BM][LDSTR];
    __shared__ __attribute__((aligned(16))) _Float16 Bs[2][BN][LDSTR];

    const int tid    = threadIdx.x;
    const int lane   = tid & 31;
    const int wave   = tid >> 5;            // 0..3
    const int l15    = lane & 15;
    const int hiHalf = lane >> 4;           // 0 or 1 (lane group)
    const int blockN = blockIdx.x * BN;
    const int blockM = blockIdx.y * BM;

    v8f acc[8] = {};
    v8h sa[4];   // staged A tile: 64x64 halfs = 512 v8h chunks / 128 thr = 4
    v8h sb[8];   // staged B tile: 128x64 halfs = 1024 chunks / 128 thr = 8

    // ---- prologue: stage tile 0 into buffer 0 ----
    #pragma unroll
    for (int i = 0; i < 4; ++i) {
        int chk = tid + i * 128;
        int r = chk >> 3, c8 = (chk & 7) * 8;
        sa[i] = *(const v8h*)(qh + (size_t)(blockM + r) * NV + c8);
    }
    #pragma unroll
    for (int i = 0; i < 8; ++i) {
        int chk = tid + i * 128;
        int r = chk >> 3, c8 = (chk & 7) * 8;
        sb[i] = *(const v8h*)(bh + (size_t)(blockN + r) * NV + c8);
    }
    #pragma unroll
    for (int i = 0; i < 4; ++i) {
        int chk = tid + i * 128;
        int r = chk >> 3, c8 = (chk & 7) * 8;
        *(v8h*)&As[0][r][c8] = sa[i];
    }
    #pragma unroll
    for (int i = 0; i < 8; ++i) {
        int chk = tid + i * 128;
        int r = chk >> 3, c8 = (chk & 7) * 8;
        *(v8h*)&Bs[0][r][c8] = sb[i];
    }
    __syncthreads();

    int cur = 0;
    for (int kt = 0; kt < NV; kt += BK) {
        const bool more = (kt + BK) < NV;    // uniform across block

        // ---- issue global loads for next tile (latency hidden by WMMAs) ----
        if (more) {
            #pragma unroll
            for (int i = 0; i < 4; ++i) {
                int chk = tid + i * 128;
                int r = chk >> 3, c8 = (chk & 7) * 8;
                sa[i] = *(const v8h*)(qh + (size_t)(blockM + r) * NV + kt + BK + c8);
            }
            #pragma unroll
            for (int i = 0; i < 8; ++i) {
                int chk = tid + i * 128;
                int r = chk >> 3, c8 = (chk & 7) * 8;
                sb[i] = *(const v8h*)(bh + (size_t)(blockN + r) * NV + kt + BK + c8);
            }
        }

        // ---- compute: 2 K-chunks of 32; preload fragments, then WMMA chain ----
        #pragma unroll
        for (int kk = 0; kk < BK; kk += 32) {
            // A fragment (16x32, ISA 7.12.2 16-bit A layout):
            //   lanes 0-15:  M=l15, K 0..7 then 16..23
            //   lanes 16-31: M=l15, K 8..15 then 24..31
            const _Float16* arow = &As[cur][wave * 16 + l15][kk];
            v8h alo = *(const v8h*)(arow + hiHalf * 8);
            v8h ahi = *(const v8h*)(arow + 16 + hiHalf * 8);
            v16h a;
            #pragma unroll
            for (int i = 0; i < 8; ++i) { a[i] = alo[i]; a[8 + i] = ahi[i]; }

            // Preload all 8 B fragments (32x16 each: lane holds N=l15,
            // K-half by lane group) into distinct registers so the 16
            // ds_load_b128 issue as a batch and the WMMAs run back-to-back.
            v16h bfr[8];
            #pragma unroll
            for (int j = 0; j < 8; ++j) {
                const _Float16* brow = &Bs[cur][j * 16 + l15][kk];
                v8h b0 = *(const v8h*)(brow + hiHalf * 8);
                v8h b1 = *(const v8h*)(brow + 16 + hiHalf * 8);
                #pragma unroll
                for (int i = 0; i < 8; ++i) { bfr[j][i] = b0[i]; bfr[j][8 + i] = b1[i]; }
            }

            #pragma unroll
            for (int j = 0; j < 8; ++j) {
                acc[j] = __builtin_amdgcn_wmma_f32_16x16x32_f16(
                    false, a, false, bfr[j], (short)0, acc[j], false, false);
            }
        }

        // ---- store staged tile into the other buffer ----
        if (more) {
            #pragma unroll
            for (int i = 0; i < 4; ++i) {
                int chk = tid + i * 128;
                int r = chk >> 3, c8 = (chk & 7) * 8;
                *(v8h*)&As[cur ^ 1][r][c8] = sa[i];
            }
            #pragma unroll
            for (int i = 0; i < 8; ++i) {
                int chk = tid + i * 128;
                int r = chk >> 3, c8 = (chk & 7) * 8;
                *(v8h*)&Bs[cur ^ 1][r][c8] = sb[i];
            }
        }
        __syncthreads();
        cur ^= 1;
    }

    // C/D layout: VGPR r -> M=r (lanes 0-15) or M=8+r (lanes 16-31); N=l15.
    const int rowBase = blockM + wave * 16 + hiHalf * 8;
    #pragma unroll
    for (int j = 0; j < 8; ++j) {
        int col = blockN + j * 16 + l15;
        #pragma unroll
        for (int r = 0; r < 8; ++r) {
            out[(size_t)(rowBase + r) * NC + col] = acc[j][r];
        }
    }
}

extern "C" void kernel_launch(void* const* d_in, const int* in_sizes, int n_in,
                              void* d_out, int out_size, void* d_ws, size_t ws_size,
                              hipStream_t stream) {
    const float* d_queries = (const float*)d_in[0];   // [NQ, NV] fp32
    const float* d_corpus  = (const float*)d_in[1];   // [NC, NV] fp32
    float* d_scores = (float*)d_out;                  // [NQ, NC] fp32

    // Workspace layout: corpus f16 | queries f16 | df counters
    char* ws = (char*)d_ws;
    _Float16* corpus_h  = (_Float16*)ws;
    _Float16* queries_h = (_Float16*)(ws + (size_t)NC * NV * sizeof(_Float16));
    int*      df        = (int*)(ws + (size_t)NC * NV * sizeof(_Float16)
                                    + (size_t)NQ * NV * sizeof(_Float16));

    k_zero_df<<<(NV + 255) / 256, 256, 0, stream>>>(df);
    k_df_convert<<<dim3(NV / 256, NC / 64), 256, 0, stream>>>(d_corpus, corpus_h, df);
    k_build_q<<<dim3(NV / 256, NQ), 256, 0, stream>>>(d_queries, df, queries_h);
    k_gemm<<<dim3(NC / BN, NQ / BM), 128, 0, stream>>>(queries_h, corpus_h, d_scores);
}
